// MyRNN_85177791414628
// MI455X (gfx1250) — matrix operand order repaired and checked by reading, not compiled, not run on hardware
//
#include <hip/hip_runtime.h>

// MI455X / gfx1250, wave32. Fused 2-layer LSTM (H=16) over T=1024, B=4096.
// One wave per 16-batch tile; weights preloaded as constant WMMA A-fragments;
// per step: 4x v_wmma_f32_16x16x32_f16 per layer; all state in registers.
// Activations use the gfx1250 hardware v_tanh_f32 (sigmoid via tanh identity).

#define H    16
#define TLEN 1024
#define NB   4096

typedef _Float16 v16h  __attribute__((ext_vector_type(16)));
typedef __fp16   v2fp  __attribute__((ext_vector_type(2)));
typedef float    v8f   __attribute__((ext_vector_type(8)));

union Frag16 { v16h h; unsigned u[8]; };

static __device__ __forceinline__ unsigned pack2h(float a, float b) {
#if __has_builtin(__builtin_amdgcn_cvt_pkrtz)
    union { v2fp h; unsigned u; } p;
    p.h = __builtin_amdgcn_cvt_pkrtz(a, b);   // v_cvt_pk_rtz_f16_f32: 1 instr
    return p.u;
#else
    union { _Float16 h[2]; unsigned u; } p;
    p.h[0] = (_Float16)a; p.h[1] = (_Float16)b;
    return p.u;
#endif
}

// gfx1250 hardware tanh (transcendental VALU op) -> 1 instruction
static __device__ __forceinline__ float htanh(float x) {
#if __has_builtin(__builtin_amdgcn_tanhf)
    return __builtin_amdgcn_tanhf(x);
#else
    float r;
    asm("v_tanh_f32 %0, %1" : "=v"(r) : "v"(x));
    return r;
#endif
}

// sigmoid(x) = 0.5*tanh(0.5*x) + 0.5  -> mul + tanh + fma (no division)
static __device__ __forceinline__ float sigm(float x) {
    return __builtin_fmaf(0.5f, htanh(0.5f * x), 0.5f);
}

static __device__ __forceinline__ v8f zero8() {
    v8f z;
#pragma unroll
    for (int i = 0; i < 8; ++i) z[i] = 0.0f;
    return z;
}
static __device__ __forceinline__ v8f wmma16(v16h a, v16h b, v8f c) {
    // D = A(16x32 f16) * B(32x16 f16) + C(16x16 f32)
    return __builtin_amdgcn_wmma_f32_16x16x32_f16(
        false, a, false, b, (short)0, c, false, false);
}

__global__ __launch_bounds__(32)
void lstm2_fused_kernel(const float* __restrict__ x,
                        const float* __restrict__ Wih0, const float* __restrict__ Whh0,
                        const float* __restrict__ bih0, const float* __restrict__ bhh0,
                        const float* __restrict__ Wih1, const float* __restrict__ Whh1,
                        const float* __restrict__ bih1, const float* __restrict__ bhh1,
                        const float* __restrict__ Wout, const float* __restrict__ bout,
                        float* __restrict__ out)
{
    const int lane = threadIdx.x;        // 0..31 (wave32)
    const int col  = lane & 15;          // batch column within tile (N index)
    const bool hi  = lane >= 16;
    const int base = hi ? 8 : 0;         // C/D layout: this lane holds rows base..base+7
    const int b0   = blockIdx.x * 16;    // batch tile start

    // ---------------- Constant A fragments (per-lane row m = col) -------------
    // 16-bit A 16x32 layout: lanes 0-15 hold K={0..7,16..23}, lanes 16-31 K={8..15,24..31}
    Frag16 A1[4], A2[4];
#pragma unroll
    for (int g = 0; g < 4; ++g) {
        const int row = g * 16 + col;    // PyTorch gate order i,f,g,o
        if (!hi) {
#pragma unroll
            for (int v = 0; v < 4; ++v)
                A1[g].u[v] = pack2h(Whh0[row * H + 2 * v], Whh0[row * H + 2 * v + 1]);
            const float bsum = bih0[row] + bhh0[row];
            A1[g].u[4] = pack2h(bsum, Wih0[row]);   // K=16 -> bias row, K=17 -> x row
            A1[g].u[5] = 0u; A1[g].u[6] = 0u; A1[g].u[7] = 0u;
#pragma unroll
            for (int v = 0; v < 4; ++v) {
                A2[g].u[v]     = pack2h(Wih1[row * H + 2 * v], Wih1[row * H + 2 * v + 1]);
                A2[g].u[4 + v] = pack2h(Whh1[row * H + 2 * v], Whh1[row * H + 2 * v + 1]);
            }
        } else {
#pragma unroll
            for (int v = 0; v < 4; ++v) {
                A1[g].u[v]     = pack2h(Whh0[row * H + 8 + 2 * v], Whh0[row * H + 9 + 2 * v]);
                A1[g].u[4 + v] = 0u;                           // K=24..31 unused in layer 1
                A2[g].u[v]     = pack2h(Wih1[row * H + 8 + 2 * v], Wih1[row * H + 9 + 2 * v]);
                A2[g].u[4 + v] = pack2h(Whh1[row * H + 8 + 2 * v], Whh1[row * H + 9 + 2 * v]);
            }
        }
    }

    // Layer-2 bias as accumulator-init fragment (C/D layout)
    v8f bias2[4];
#pragma unroll
    for (int g = 0; g < 4; ++g)
#pragma unroll
        for (int r = 0; r < 8; ++r)
            bias2[g][r] = bih1[g * 16 + base + r] + bhh1[g * 16 + base + r];

    // ---------------- Recurrent state (all in registers) ----------------------
    float c1[8], c2[8], h2f[8];
    unsigned p1[4], q1[4], p2[4], q2[4];   // packed-f16 h halves (own / partner)
#pragma unroll
    for (int r = 0; r < 8; ++r) { c1[r] = 0.f; c2[r] = 0.f; h2f[r] = 0.f; }
#pragma unroll
    for (int j = 0; j < 4; ++j) { p1[j] = 0u; q1[j] = 0u; p2[j] = 0u; q2[j] = 0u; }

    const float4* __restrict__ xrow =
        (const float4*)(x + (size_t)(b0 + col) * TLEN);

    for (int t0 = 0; t0 < TLEN / 4; ++t0) {
        const float4 xq = xrow[t0];
#pragma unroll
        for (int tt = 0; tt < 4; ++tt) {
            const float xv = (&xq.x)[tt];

            // ---- Layer 1: B = [h1(t-1) ; 1, x, 0...] ----
            Frag16 B1;
#pragma unroll
            for (int j = 0; j < 4; ++j) {
                B1.u[j]     = p1[j];                      // K=0..7 / 8..15 (own rows)
                B1.u[4 + j] = 0u;                         // K=16..23 / 24..31
            }
            if (!hi) B1.u[4] = pack2h(1.0f, xv);          // K=16 -> 1.0, K=17 -> x

            v8f g0 = wmma16(A1[0].h, B1.h, zero8());
            v8f g1 = wmma16(A1[1].h, B1.h, zero8());
            v8f g2 = wmma16(A1[2].h, B1.h, zero8());
            v8f g3 = wmma16(A1[3].h, B1.h, zero8());

            float h1f[8];
#pragma unroll
            for (int r = 0; r < 8; ++r) {
                const float ig = sigm(g0[r]);
                const float fg = sigm(g1[r]);
                const float gg = htanh(g2[r]);
                const float og = sigm(g3[r]);
                c1[r]  = __builtin_fmaf(fg, c1[r], ig * gg);
                h1f[r] = og * htanh(c1[r]);
            }
#pragma unroll
            for (int j = 0; j < 4; ++j) {
                p1[j] = pack2h(h1f[2 * j], h1f[2 * j + 1]);
                q1[j] = __shfl_xor(p1[j], 16, 32);         // swap column halves
            }

            // ---- Layer 2: B = [h1(t) ; h2(t-1)], full K=32 ----
            Frag16 B2;
#pragma unroll
            for (int j = 0; j < 4; ++j) {
                B2.u[j]     = p1[j];                       // K 0..7 / 8..15 (h1)
                B2.u[4 + j] = p2[j];                       // K 16..23 / 24..31 (h2)
            }

            v8f a0 = wmma16(A2[0].h, B2.h, bias2[0]);
            v8f a1 = wmma16(A2[1].h, B2.h, bias2[1]);
            v8f a2 = wmma16(A2[2].h, B2.h, bias2[2]);
            v8f a3 = wmma16(A2[3].h, B2.h, bias2[3]);

#pragma unroll
            for (int r = 0; r < 8; ++r) {
                const float ig = sigm(a0[r]);
                const float fg = sigm(a1[r]);
                const float gg = htanh(a2[r]);
                const float og = sigm(a3[r]);
                c2[r]  = __builtin_fmaf(fg, c2[r], ig * gg);
                h2f[r] = og * htanh(c2[r]);
            }
#pragma unroll
            for (int j = 0; j < 4; ++j) {
                p2[j] = pack2h(h2f[2 * j], h2f[2 * j + 1]);
                q2[j] = __shfl_xor(p2[j], 16, 32);
            }
        }
    }

    // ---------------- Head: relu(hT) @ Wout.T + bout --------------------------
    float part[5];
#pragma unroll
    for (int o = 0; o < 5; ++o) {
        float s = 0.0f;
#pragma unroll
        for (int r = 0; r < 8; ++r) {
            const float hv = h2f[r] > 0.0f ? h2f[r] : 0.0f;
            s += hv * Wout[o * H + base + r];
        }
        s += __shfl_xor(s, 16, 32);      // combine row halves of this column
        part[o] = s;
    }
    if (!hi) {
#pragma unroll
        for (int o = 0; o < 5; ++o)
            out[(size_t)(b0 + col) * 5 + o] = part[o] + bout[o];
    }
}

extern "C" void kernel_launch(void* const* d_in, const int* in_sizes, int n_in,
                              void* d_out, int out_size, void* d_ws, size_t ws_size,
                              hipStream_t stream)
{
    (void)in_sizes; (void)n_in; (void)out_size; (void)d_ws; (void)ws_size;
    const float* x    = (const float*)d_in[0];
    const float* Wih0 = (const float*)d_in[1];
    const float* Whh0 = (const float*)d_in[2];
    const float* bih0 = (const float*)d_in[3];
    const float* bhh0 = (const float*)d_in[4];
    const float* Wih1 = (const float*)d_in[5];
    const float* Whh1 = (const float*)d_in[6];
    const float* bih1 = (const float*)d_in[7];
    const float* bhh1 = (const float*)d_in[8];
    const float* Wout = (const float*)d_in[9];
    const float* bout = (const float*)d_in[10];
    float* out = (float*)d_out;

    // One wave (32 threads) per 16-batch tile: 4096/16 = 256 single-wave workgroups.
    lstm2_fused_kernel<<<NB / 16, 32, 0, stream>>>(
        x, Wih0, Whh0, bih0, bhh0, Wih1, Whh1, bih1, bhh1, Wout, bout, out);
}
